// PointerGeneratorNetwork_7146825580857
// MI455X (gfx1250) — matrix operand order repaired and checked by reading, not compiled, not run on hardware
//
#include <hip/hip_runtime.h>
#include <hip/hip_bf16.h>

// ---------------- problem constants (from reference) ----------------
constexpr int kS = 400, kB = 32, kT = 32;
constexpr int kV = 50000, kOOV = 50, kE = 128, kH = 256;
constexpr int kG = 4 * kH;          // 1024 gate width
constexpr int k2H = 2 * kH;         // 512

typedef __attribute__((ext_vector_type(16))) _Float16 v16h;
typedef __attribute__((ext_vector_type(8)))  _Float16 v8h;
typedef __attribute__((ext_vector_type(8)))  float    v8f;

__device__ __forceinline__ float sigf(float x) { return 1.0f / (1.0f + __expf(-x)); }

__device__ __forceinline__ float wave_reduce_sum(float v) {
  #pragma unroll
  for (int m = 16; m; m >>= 1) v += __shfl_xor(v, m, 32);
  return v;
}

// ---- WMMA fragment loaders (CDNA5 16x16x32 f16 layouts, ISA 7.12.2) ----
// A: row-major [M][K] f16. Lanes 0-15 rows, K halves {kb..kb+7, kb+16..kb+23}, kb=0/8.
__device__ __forceinline__ v16h load_A_frag(const _Float16* __restrict__ A, int lda,
                                            int m0, int k0, int lane) {
  int row = m0 + (lane & 15);
  int kb  = k0 + ((lane >> 4) << 3);
  const _Float16* p = A + (size_t)row * lda + kb;
  v8h lo = *(const v8h*)(p);
  v8h hi = *(const v8h*)(p + 16);
  v16h r;
  #pragma unroll
  for (int q = 0; q < 8; ++q) { r[q] = lo[q]; r[8 + q] = hi[q]; }
  return r;
}
// B from N-major weights W[N][K] (reference stores (out,in) row-major => already N-major).
// lane = column n0+(lane&15); K halves {kb..kb+7, kb+16..kb+23}, kb=k0+0/8 (same as A).
__device__ __forceinline__ v16h load_B_frag(const _Float16* __restrict__ W, int ldw,
                                            int n0, int k0, int lane) {
  int n  = n0 + (lane & 15);
  int kb = k0 + ((lane >> 4) << 3);
  const _Float16* p = W + (size_t)n * ldw + kb;
  v8h lo = *(const v8h*)(p);
  v8h hi = *(const v8h*)(p + 16);
  v16h r;
  #pragma unroll
  for (int q = 0; q < 8; ++q) { r[q] = lo[q]; r[8 + q] = hi[q]; }
  return r;
}

// ---------------- generic WMMA GEMM: C = A(MxK f16) * W(NxK f16)^T (+bias[+bias2]) ----------------
__global__ void __launch_bounds__(256)
k_gemm(const _Float16* __restrict__ A, int lda,
       const _Float16* __restrict__ W, int ldw,
       float* __restrict__ C, int ldc,
       _Float16* __restrict__ Ch, int ldch,
       const float* __restrict__ bias, const float* __restrict__ bias2,
       int M, int N, int K, int act) {
  int wave = (blockIdx.x << 3) + (threadIdx.x >> 5);
  int lane = threadIdx.x & 31;
  int ntn = (N + 15) >> 4;
  int tm = wave / ntn, tn = wave - tm * ntn;
  if (tm * 16 >= M) return;            // wave-uniform exit (EXEC all-ones for WMMA)
  int m0 = tm << 4, n0 = tn << 4;
  v8f acc = {};
  for (int k0 = 0; k0 < K; k0 += 32) {
    v16h a = load_A_frag(A, lda, m0, k0, lane);
    v16h b = load_B_frag(W, ldw, n0, k0, lane);
    acc = __builtin_amdgcn_wmma_f32_16x16x32_f16(false, a, false, b, (short)0, acc, false, false);
  }
  int col = n0 + (lane & 15);
  float bv = 0.f;
  if (bias)  bv += bias[col];
  if (bias2) bv += bias2[col];
  int rb = m0 + ((lane >> 4) << 3);
  #pragma unroll
  for (int r = 0; r < 8; ++r) {
    int row = rb + r;
    if (row < M && col < N) {
      float v = acc[r] + bv;
      if (act == 1) v = fmaxf(v, 0.f);
      C[(size_t)row * ldc + col] = v;
      if (Ch) Ch[(size_t)row * ldch + col] = (_Float16)v;
    }
  }
}

// ---------------- helpers ----------------
__global__ void k_f32_to_f16(const float* __restrict__ s, _Float16* __restrict__ d, int n) {
  int i = blockIdx.x * blockDim.x + threadIdx.x;
  if (i < n) d[i] = (_Float16)s[i];
}

__global__ void k_embed_f16(const float* __restrict__ emb, const int* __restrict__ tok,
                            _Float16* __restrict__ out, int ntok) {
  int i = blockIdx.x * blockDim.x + threadIdx.x;
  if (i >= ntok * kE) return;
  int r = i >> 7, e = i & (kE - 1);
  out[i] = (_Float16)emb[(size_t)tok[r] * kE + e];
}

// Wd_cat[n][k] = k<128 ? dec_Wih[n][k] : dec_Whh[n][k-128]  (N=1024, K=384)
__global__ void k_pack_wdcat(const float* __restrict__ Wih, const float* __restrict__ Whh,
                             _Float16* __restrict__ out) {
  int i = blockIdx.x * blockDim.x + threadIdx.x;
  if (i >= kG * (kE + kH)) return;
  int n = i / (kE + kH), k = i - n * (kE + kH);
  float v = (k < kE) ? Wih[n * kE + k] : Whh[n * kH + (k - kE)];
  out[i] = (_Float16)v;
}

// ---------------- persistent bidirectional LSTM scan (2 blocks = fwd/bwd) ----------------
__global__ void __launch_bounds__(1024)
k_enc_scan(const float* __restrict__ Xf, const float* __restrict__ Xb,
           const _Float16* __restrict__ Whhf, const _Float16* __restrict__ Whhb,
           float* __restrict__ enc_out, _Float16* __restrict__ enc_out_h,
           _Float16* __restrict__ hcat, _Float16* __restrict__ ccat) {
  const int dir = blockIdx.x;
  const float*    Xpre = dir ? Xb : Xf;
  const _Float16* Whh  = dir ? Whhb : Whhf;
  const int dofs = dir ? kH : 0;
  extern __shared__ char smem[];
  float*    g_s = (float*)smem;                                   // 32x1024 f32
  float*    c_s = (float*)(smem + 32 * kG * sizeof(float));       // 32x256  f32
  _Float16* h_s = (_Float16*)(smem + 32 * kG * 4 + 32 * kH * 4);  // 32x256  f16
  int tid = threadIdx.x, wave = tid >> 5, lane = tid & 31;
  for (int i = tid; i < 32 * kH; i += 1024) { c_s[i] = 0.f; h_s[i] = (_Float16)0.f; }
  __syncthreads();
  for (int s = 0; s < kS; ++s) {
    int t = dir ? (kS - 1 - s) : s;
    const float* Xt = Xpre + (size_t)t * (32 * kG);
    // recurrent GEMM: g = h @ Whh^T ; 2x64 tiles over 32 waves (A from LDS)
    for (int tile = wave; tile < 128; tile += 32) {
      int m0 = (tile >> 6) << 4;
      int n0 = (tile & 63) << 4;
      int row = m0 + (lane & 15);
      int kb  = (lane >> 4) << 3;
      v8f acc = {};
      for (int k0 = 0; k0 < kH; k0 += 32) {
        const _Float16* pa = h_s + row * kH + k0 + kb;
        v8h lo = *(const v8h*)pa;
        v8h hi = *(const v8h*)(pa + 16);
        v16h afrag;
        #pragma unroll
        for (int q = 0; q < 8; ++q) { afrag[q] = lo[q]; afrag[8 + q] = hi[q]; }
        v16h bfrag = load_B_frag(Whh, kH, n0, k0, lane);
        acc = __builtin_amdgcn_wmma_f32_16x16x32_f16(false, afrag, false, bfrag,
                                                     (short)0, acc, false, false);
      }
      int col = n0 + (lane & 15);
      int rb = m0 + ((lane >> 4) << 3);
      #pragma unroll
      for (int r = 0; r < 8; ++r)
        g_s[(rb + r) * kG + col] = acc[r] + Xt[(rb + r) * kG + col];
    }
    __syncthreads();
    // pointwise gates
    for (int i = tid; i < 32 * kH; i += 1024) {
      int b = i >> 8, u = i & (kH - 1);
      const float* gr = g_s + b * kG;
      float ci = sigf(gr[kH + u]) * c_s[i] + sigf(gr[u]) * tanhf(gr[2 * kH + u]);
      float hv = sigf(gr[3 * kH + u]) * tanhf(ci);
      c_s[i] = ci; h_s[i] = (_Float16)hv;
      size_t o = ((size_t)t * 32 + b) * k2H + dofs + u;
      enc_out[o] = hv; enc_out_h[o] = (_Float16)hv;
    }
    __syncthreads();
  }
  for (int i = tid; i < 32 * kH; i += 1024) {
    int b = i >> 8, u = i & (kH - 1);
    hcat[b * k2H + dofs + u] = h_s[i];
    ccat[b * k2H + dofs + u] = (_Float16)c_s[i];
  }
}

// ---------------- decoder small kernels ----------------
__global__ void k_pack_ax(const float* __restrict__ ctx, const _Float16* __restrict__ embtok,
                          _Float16* __restrict__ Ax) {
  int i = blockIdx.x * blockDim.x + threadIdx.x;
  if (i >= 32 * (k2H + kE)) return;
  int b = i / (k2H + kE), j = i - b * (k2H + kE);
  Ax[i] = (j < k2H) ? (_Float16)ctx[b * k2H + j] : embtok[b * kE + (j - k2H)];
}

__global__ void k_dec_point(const float* __restrict__ g, float* __restrict__ c_dec,
                            _Float16* __restrict__ Ag, _Float16* __restrict__ Ao1,
                            float* __restrict__ dh32) {
  int i = blockIdx.x * blockDim.x + threadIdx.x;
  if (i >= 32 * kH) return;
  int b = i >> 8, u = i & (kH - 1);
  const float* gr = g + b * kG;
  float c = sigf(gr[kH + u]) * c_dec[i] + sigf(gr[u]) * tanhf(gr[2 * kH + u]);
  float h = sigf(gr[3 * kH + u]) * tanhf(c);
  c_dec[i] = c;
  Ag[b * (kE + kH) + kE + u] = (_Float16)h;     // h state for next gates GEMM
  Ao1[b * 2 * k2H + u]       = (_Float16)h;     // dh = [h, c] f16
  Ao1[b * 2 * k2H + kH + u]  = (_Float16)c;
  dh32[b * k2H + u]          = h;               // dh f32 for p_gen
  dh32[b * k2H + kH + u]     = c;
}

__global__ void __launch_bounds__(1024)
k_attn_e(const float* __restrict__ enc_feat, const float* __restrict__ af,
         const float* __restrict__ wv, float* __restrict__ e) {
  int s = blockIdx.x, b = threadIdx.x >> 5, lane = threadIdx.x & 31;
  const float* ef  = enc_feat + ((size_t)s * 32 + b) * k2H;
  const float* afb = af + b * k2H;
  float acc = 0.f;
  for (int d = lane; d < k2H; d += 32) acc += tanhf(ef[d] + afb[d]) * wv[d];
  acc = wave_reduce_sum(acc);
  if (lane == 0) e[s * 32 + b] = acc;
}

__global__ void __launch_bounds__(512)
k_attn_softmax(const float* __restrict__ e, const int* __restrict__ inputs,
               float* __restrict__ a, float* __restrict__ atts_t) {
  __shared__ float red[512];
  int b = blockIdx.x, tid = threadIdx.x;
  float val = 0.f, m = 0.f, ev = -1e30f;
  if (tid < kS) { val = e[tid * 32 + b]; m = (inputs[tid * 32 + b] > 0) ? 1.f : 0.f; ev = val; }
  red[tid] = ev; __syncthreads();
  for (int st = 256; st; st >>= 1) { if (tid < st) red[tid] = fmaxf(red[tid], red[tid + st]); __syncthreads(); }
  float mx = red[0]; __syncthreads();
  float ex = (tid < kS) ? __expf(val - mx) * m : 0.f;
  red[tid] = ex; __syncthreads();
  for (int st = 256; st; st >>= 1) { if (tid < st) red[tid] += red[tid + st]; __syncthreads(); }
  float sum = red[0];
  if (tid < kS) { float av = ex / sum; a[tid * 32 + b] = av; atts_t[tid * 32 + b] = av; }
}

__global__ void __launch_bounds__(512)
k_context(const float* __restrict__ a, const float* __restrict__ enc_out,
          float* __restrict__ ctx, _Float16* __restrict__ Ao1) {
  __shared__ float as[kS];
  int b = blockIdx.x, tid = threadIdx.x;
  if (tid < kS) as[tid] = a[tid * 32 + b];
  __syncthreads();
  const float* eo = enc_out + b * k2H + tid;
  float acc = 0.f;
  for (int s = 0; s < kS; ++s) acc += as[s] * eo[(size_t)s * 32 * k2H];
  ctx[b * k2H + tid] = acc;
  Ao1[b * 2 * k2H + k2H + tid] = (_Float16)acc;  // [dh | context] f16
}

__global__ void __launch_bounds__(1024)
k_pgen(const float* __restrict__ ctx, const float* __restrict__ dh32,
       const float* __restrict__ x32, const float* __restrict__ Wpg,
       const float* __restrict__ bpg, float* __restrict__ pgen) {
  int b = threadIdx.x >> 5, lane = threadIdx.x & 31;
  float acc = 0.f;
  for (int j = lane; j < 2 * k2H + kE; j += 32) {
    float v = (j < k2H) ? ctx[b * k2H + j]
            : (j < 2 * k2H) ? dh32[b * k2H + (j - k2H)]
            : x32[b * kE + (j - 2 * k2H)];
    acc += v * Wpg[j];
  }
  acc = wave_reduce_sum(acc);
  if (lane == 0) pgen[b] = sigf(acc + bpg[0]);
}

__global__ void __launch_bounds__(1024)
k_vocab(const float* __restrict__ logits, const float* __restrict__ pgen,
        const float* __restrict__ a, const int* __restrict__ tex,
        float* __restrict__ out_t) {
  __shared__ float red[1024];
  int b = blockIdx.x, tid = threadIdx.x;
  const float* lb = logits + (size_t)b * kV;
  float* ob = out_t + (size_t)b * (kV + kOOV);
  float mx = -1e30f;
  for (int v = tid; v < kV; v += 1024) mx = fmaxf(mx, lb[v]);
  red[tid] = mx; __syncthreads();
  for (int st = 512; st; st >>= 1) { if (tid < st) red[tid] = fmaxf(red[tid], red[tid + st]); __syncthreads(); }
  mx = red[0]; __syncthreads();
  float sum = 0.f;
  for (int v = tid; v < kV; v += 1024) sum += __expf(lb[v] - mx);
  red[tid] = sum; __syncthreads();
  for (int st = 512; st; st >>= 1) { if (tid < st) red[tid] += red[tid + st]; __syncthreads(); }
  sum = red[0];
  float pg = pgen[b], inv = pg / sum;
  for (int v = tid; v < kV; v += 1024) ob[v] = __expf(lb[v] - mx) * inv;
  if (tid < kOOV) ob[kV + tid] = 0.f;
  __syncthreads();
  float cpg = 1.f - pg;
  for (int s = tid; s < kS; s += 1024)
    atomicAdd(&ob[tex[s * 32 + b]], cpg * a[s * 32 + b]);
}

// ---------------- host orchestration ----------------
extern "C" void kernel_launch(void* const* d_in, const int* in_sizes, int n_in,
                              void* d_out, int out_size, void* d_ws, size_t ws_size,
                              hipStream_t stream) {
  const int*   inputs     = (const int*)d_in[0];
  const int*   out_tokens = (const int*)d_in[2];
  const int*   tex        = (const int*)d_in[3];
  const float* embedding  = (const float*)d_in[5];
  const float* eWihf = (const float*)d_in[6],  *eWhhf = (const float*)d_in[7];
  const float* ebihf = (const float*)d_in[8],  *ebhhf = (const float*)d_in[9];
  const float* eWihb = (const float*)d_in[10], *eWhhb = (const float*)d_in[11];
  const float* ebihb = (const float*)d_in[12], *ebhhb = (const float*)d_in[13];
  const float* Wfeat = (const float*)d_in[14];
  const float* Wrh = (const float*)d_in[15], *brh = (const float*)d_in[16];
  const float* Wrc = (const float*)d_in[17], *brc = (const float*)d_in[18];
  const float* Waf = (const float*)d_in[19], *baf = (const float*)d_in[20];
  const float* wv  = (const float*)d_in[21];
  const float* Wctx = (const float*)d_in[22], *bctx = (const float*)d_in[23];
  const float* dWih = (const float*)d_in[24], *dWhh = (const float*)d_in[25];
  const float* dbih = (const float*)d_in[26], *dbhh = (const float*)d_in[27];
  const float* Wpg = (const float*)d_in[28], *bpg = (const float*)d_in[29];
  const float* Wo1 = (const float*)d_in[30], *bo1 = (const float*)d_in[31];
  const float* Wo2 = (const float*)d_in[32], *bo2 = (const float*)d_in[33];
  (void)in_sizes; (void)n_in; (void)out_size; (void)ws_size;

  size_t off = 0;
  auto take = [&](size_t bytes) -> char* {
    char* p = (char*)d_ws + off;
    off = (off + bytes + 255) & ~(size_t)255;
    return p;
  };
  // f16 weights
  _Float16* hWihf = (_Float16*)take((size_t)kG * kE * 2);
  _Float16* hWhhf = (_Float16*)take((size_t)kG * kH * 2);
  _Float16* hWihb = (_Float16*)take((size_t)kG * kE * 2);
  _Float16* hWhhb = (_Float16*)take((size_t)kG * kH * 2);
  _Float16* hWfeat = (_Float16*)take((size_t)k2H * k2H * 2);
  _Float16* hWrh = (_Float16*)take((size_t)kH * k2H * 2);
  _Float16* hWrc = (_Float16*)take((size_t)kH * k2H * 2);
  _Float16* hWaf = (_Float16*)take((size_t)k2H * k2H * 2);
  _Float16* hWctx = (_Float16*)take((size_t)kE * (k2H + kE) * 2);
  _Float16* hWdcat = (_Float16*)take((size_t)kG * (kE + kH) * 2);
  _Float16* hWo1 = (_Float16*)take((size_t)kH * 2 * k2H * 2);
  _Float16* hWo2 = (_Float16*)take((size_t)kV * kH * 2);
  // activations
  _Float16* emb_in  = (_Float16*)take((size_t)kS * kB * kE * 2);
  _Float16* emb_dec = (_Float16*)take((size_t)kT * kB * kE * 2);
  float* Xf = (float*)take((size_t)kS * kB * kG * 4);
  float* Xb = (float*)take((size_t)kS * kB * kG * 4);
  float*    enc_out   = (float*)take((size_t)kS * kB * k2H * 4);
  _Float16* enc_out_h = (_Float16*)take((size_t)kS * kB * k2H * 2);
  float*    enc_feat  = (float*)take((size_t)kS * kB * k2H * 4);
  _Float16* hcat = (_Float16*)take((size_t)kB * k2H * 2);
  _Float16* ccat = (_Float16*)take((size_t)kB * k2H * 2);
  float* tmp32 = (float*)take((size_t)kB * kG * 4);
  float* c_dec = (float*)take((size_t)kB * kH * 4);
  float* ctx   = (float*)take((size_t)kB * k2H * 4);
  _Float16* Ax  = (_Float16*)take((size_t)kB * (k2H + kE) * 2);
  _Float16* Ag  = (_Float16*)take((size_t)kB * (kE + kH) * 2);
  _Float16* Ao1 = (_Float16*)take((size_t)kB * 2 * k2H * 2);
  float* x32  = (float*)take((size_t)kB * kE * 4);
  float* g    = (float*)take((size_t)kB * kG * 4);
  float* dh32 = (float*)take((size_t)kB * k2H * 4);
  float* af   = (float*)take((size_t)kB * k2H * 4);
  float* ebuf = (float*)take((size_t)kS * kB * 4);
  float* abuf = (float*)take((size_t)kS * kB * 4);
  float* pgen = (float*)take(kB * 4);
  float*    hid   = (float*)take((size_t)kB * kH * 4);
  _Float16* hid_h = (_Float16*)take((size_t)kB * kH * 2);
  float* logits = (float*)take((size_t)kB * kV * 4);

  float* outs = (float*)d_out;                                   // (T,B,V+OOV)
  float* atts = (float*)d_out + (size_t)kT * kB * (kV + kOOV);   // (T,S,B)

  auto cvt = [&](const float* s, _Float16* d, int n) {
    k_f32_to_f16<<<(n + 255) / 256, 256, 0, stream>>>(s, d, n);
  };
  auto gemm = [&](const _Float16* A, int lda, const _Float16* W, int ldw,
                  float* C, int ldc, _Float16* Ch, int ldch,
                  const float* b1, const float* b2, int M, int N, int K, int act) {
    int tiles = ((M + 15) / 16) * ((N + 15) / 16);
    k_gemm<<<(tiles + 7) / 8, 256, 0, stream>>>(A, lda, W, ldw, C, ldc, Ch, ldch,
                                                b1, b2, M, N, K, act);
  };

  // --- setup: weight conversion + embedding gathers ---
  cvt(eWihf, hWihf, kG * kE);   cvt(eWhhf, hWhhf, kG * kH);
  cvt(eWihb, hWihb, kG * kE);   cvt(eWhhb, hWhhb, kG * kH);
  cvt(Wfeat, hWfeat, k2H * k2H); cvt(Wrh, hWrh, kH * k2H); cvt(Wrc, hWrc, kH * k2H);
  cvt(Waf, hWaf, k2H * k2H);     cvt(Wctx, hWctx, kE * (k2H + kE));
  cvt(Wo1, hWo1, kH * 2 * k2H);  cvt(Wo2, hWo2, kV * kH);
  k_pack_wdcat<<<(kG * (kE + kH) + 255) / 256, 256, 0, stream>>>(dWih, dWhh, hWdcat);
  k_embed_f16<<<(kS * kB * kE + 255) / 256, 256, 0, stream>>>(embedding, inputs, emb_in, kS * kB);
  k_embed_f16<<<(kT * kB * kE + 255) / 256, 256, 0, stream>>>(embedding, out_tokens, emb_dec, kT * kB);

  // --- hoisted input GEMMs: Xpre = emb @ Wih^T + bih + bhh (both directions) ---
  gemm(emb_in, kE, hWihf, kE, Xf, kG, nullptr, 0, ebihf, ebhhf, kS * kB, kG, kE, 0);
  gemm(emb_in, kE, hWihb, kE, Xb, kG, nullptr, 0, ebihb, ebhhb, kS * kB, kG, kE, 0);

  // --- bidirectional LSTM scan: 2 persistent blocks, 176KB dynamic LDS each ---
  size_t lds = (size_t)32 * kG * 4 + (size_t)32 * kH * 4 + (size_t)32 * kH * 2;
  k_enc_scan<<<2, 1024, lds, stream>>>(Xf, Xb, hWhhf, hWhhb, enc_out, enc_out_h, hcat, ccat);

  // --- enc_feat, decoder init states ---
  gemm(enc_out_h, k2H, hWfeat, k2H, enc_feat, k2H, nullptr, 0, nullptr, nullptr,
       kS * kB, k2H, k2H, 0);
  gemm(hcat, k2H, hWrh, k2H, tmp32, kH, Ag + kE, kE + kH, brh, nullptr, kB, kH, k2H, 1); // dh0 -> h state
  gemm(ccat, k2H, hWrc, k2H, c_dec, kH, nullptr, 0, brc, nullptr, kB, kH, k2H, 1);       // dc0 -> c state
  hipMemsetAsync(ctx, 0, (size_t)kB * k2H * 4, stream);

  // --- decoder: T sequential steps ---
  for (int t = 0; t < kT; ++t) {
    k_pack_ax<<<(kB * (k2H + kE) + 255) / 256, 256, 0, stream>>>(
        ctx, emb_dec + (size_t)t * kB * kE, Ax);
    gemm(Ax, k2H + kE, hWctx, k2H + kE, x32, kE, Ag, kE + kH, bctx, nullptr,
         kB, kE, k2H + kE, 0);                                    // x (f32 + f16 into Ag)
    gemm(Ag, kE + kH, hWdcat, kE + kH, g, kG, nullptr, 0, dbih, dbhh,
         kB, kG, kE + kH, 0);                                     // gates
    k_dec_point<<<(kB * kH + 255) / 256, 256, 0, stream>>>(g, c_dec, Ag, Ao1, dh32);
    gemm(Ao1, 2 * k2H, hWaf, k2H, af, k2H, nullptr, 0, baf, nullptr,
         kB, k2H, k2H, 0);                                        // dh @ W_af^T + b_af
    k_attn_e<<<kS, 1024, 0, stream>>>(enc_feat, af, wv, ebuf);
    k_attn_softmax<<<kB, 512, 0, stream>>>(ebuf, inputs, abuf, atts + (size_t)t * kS * kB);
    k_context<<<kB, 512, 0, stream>>>(abuf, enc_out, ctx, Ao1);
    k_pgen<<<1, 1024, 0, stream>>>(ctx, dh32, x32, Wpg, bpg, pgen);
    gemm(Ao1, 2 * k2H, hWo1, 2 * k2H, hid, kH, hid_h, kH, bo1, nullptr,
         kB, kH, 2 * k2H, 0);                                     // hid
    gemm(hid_h, kH, hWo2, kH, logits, kV, nullptr, 0, bo2, nullptr,
         kB, kV, kH, 0);                                          // big vocab GEMM
    k_vocab<<<kB, 1024, 0, stream>>>(logits, pgen, abuf, tex,
                                     outs + (size_t)t * kB * (kV + kOOV));
  }
}
